// TensorMessage_Tensors_90443421319800
// MI455X (gfx1250) — compile-verified, matrix operand order checked by reading, not run on hardware
//
#include <hip/hip_runtime.h>
#include <hip/hip_bf16.h>
#include <math.h>
#include <stdint.h>

typedef __attribute__((ext_vector_type(16))) _Float16 v16h;
typedef __attribute__((ext_vector_type(8)))  _Float16 v8h;
typedef __attribute__((ext_vector_type(8)))  float    v8f;
// gcc-style vector exactly matching the async-LDS builtin's parameter pointee type
typedef int v4i_vs __attribute__((__vector_size__(4 * sizeof(int))));

#define S_DIM  128
#define V_DIM  64
#define T_DIM  64
#define D_IN   320     // S + V + 2T
#define D_MID  160
#define K1STEPS 10     // 320/32
#define K2STEPS 5      // 160/32
#define NT1    10      // 160/16
#define NT2    4       // 64/16
#define NUM_GRAPHS 4096

// CDNA5 async copy global->LDS (ASYNCcnt path): use if this toolchain declares it.
#if defined(__has_builtin)
#  if __has_builtin(__builtin_amdgcn_global_load_async_to_lds_b128) && \
      __has_builtin(__builtin_amdgcn_s_wait_asynccnt)
#    define USE_ASYNC_LDS 1
#  endif
#endif

#define G_AS   __attribute__((address_space(1)))
#define LDS_AS __attribute__((address_space(3)))

// K index inside a 16-half WMMA fragment (16-bit A/B layout, CDNA5 ISA 7.12.2):
// lane half hi (=lane/16): j<8 -> K = hi*8 + j ; j>=8 -> K = 16 + hi*8 + (j-8)
__device__ __forceinline__ int kmap(int j, int hi) {
    return (j < 8) ? (hi * 8 + j) : (8 + hi * 8 + j);
}

// Pack a row-major f32 weight matrix W[K][Nn] into f16 B-fragments:
// out[((nt*ksteps + ks)*32 + lane)*16 + j] = W[ks*32 + kmap(j,hi)][nt*16 + lane%16]
__global__ void pack_weights_kernel(const float* __restrict__ W, _Float16* __restrict__ out,
                                    int Nn, int ksteps, int total) {
    int idx = blockIdx.x * blockDim.x + threadIdx.x;
    if (idx >= total) return;
    int j    = idx & 15;
    int lane = (idx >> 4) & 31;
    int rest = idx >> 9;
    int ks   = rest % ksteps;
    int nt   = rest / ksteps;
    int hi   = lane >> 4;
    int n    = nt * 16 + (lane & 15);
    int k    = ks * 32 + kmap(j, hi);
    out[idx] = (_Float16)W[(size_t)k * Nn + n];
}

__global__ void zero_out_kernel(float* __restrict__ o, int n) {
    int i = blockIdx.x * blockDim.x + threadIdx.x;
    if (i < n) o[i] = 0.0f;
}

__global__ __launch_bounds__(128)
void tensor_msg_kernel(const float* __restrict__ h, const float* __restrict__ v,
                       const float* __restrict__ t, const float* __restrict__ frames,
                       const long long* __restrict__ batch,
                       const _Float16* __restrict__ w1p, const float* __restrict__ b1,
                       const _Float16* __restrict__ w2p, const float* __restrict__ b2,
                       float* __restrict__ out, int Ntot) {
    __shared__ __align__(16) float     t_lds[16 * 576];          // 36 KB: t tile cache
    __shared__ __align__(16) _Float16  norms[16 * D_IN];         // 10 KB: MLP input (f16)
    __shared__ __align__(16) _Float16  hidden[16 * D_MID];       //  5 KB: SiLU output (f16)
    __shared__ float alphas_s[16 * T_DIM];                       //  4 KB
    __shared__ float partial[16 * 8 * 9];                        // 4.5 KB

    const int tid      = threadIdx.x;
    const int lane     = tid & 31;
    const int wave     = tid >> 5;
    const int col      = lane & 15;   // C column / A row index
    const int hi       = lane >> 4;
    const int nodeBase = blockIdx.x * 16;
    const int nodesIn  = Ntot - nodeBase;   // >=1; full tiles when Ntot%16==0

    // ---- Phase 1: stream t (once!), h, v into LDS; build f16 feature rows ----
    {
        const int tileF4 = ((nodesIn >= 16 ? 16 : nodesIn) * 576) >> 2;
        const float4* tg = (const float4*)(t + (size_t)nodeBase * 576);
#if USE_ASYNC_LDS
        // Direct global->LDS DMA of the t tile; bypasses VGPRs, tracked by ASYNCcnt.
        for (int i = tid; i < 2304; i += 128) {
            if (i < tileF4) {
                G_AS v4i_vs*   gp = (G_AS v4i_vs*)(uintptr_t)(tg + i);
                // generic shared addr [31:0] == LDS byte offset (aperture rule)
                LDS_AS v4i_vs* lp = (LDS_AS v4i_vs*)(uintptr_t)(uint32_t)(uintptr_t)(&t_lds[i * 4]);
                __builtin_amdgcn_global_load_async_to_lds_b128(gp, lp, 0, 0);
            }
        }
#else
        for (int i = tid; i < 2304; i += 128) {
            float4 val = (i < tileF4) ? tg[i] : make_float4(0.f, 0.f, 0.f, 0.f);
            ((float4*)t_lds)[i] = val;
        }
#endif
        for (int i = tid; i < 16 * S_DIM; i += 128) {
            int node = i >> 7, f = i & 127;
            int gn = nodeBase + node; if (gn >= Ntot) gn = Ntot - 1;
            norms[node * D_IN + f] = (_Float16)h[(size_t)gn * S_DIM + f];
        }
        for (int i = tid; i < 16 * V_DIM; i += 128) {
            int node = i >> 6, c = i & 63;
            int gn = nodeBase + node; if (gn >= Ntot) gn = Ntot - 1;
            const float* vp = v + ((size_t)gn * V_DIM + c) * 3;
            float x = vp[0], y = vp[1], z = vp[2];
            norms[node * D_IN + S_DIM + c] = (_Float16)sqrtf(x * x + y * y + z * z);
        }
    }
#if USE_ASYNC_LDS
    __builtin_amdgcn_s_wait_asynccnt(0);   // my wave's async LDS writes landed
#endif
    __syncthreads();
    // trace + Frobenius norm from the LDS copy of t
    for (int i = tid; i < 16 * T_DIM; i += 128) {
        int node = i >> 6, c = i & 63;
        const float* tc = &t_lds[node * 576 + c * 9];
        float tr = tc[0] + tc[4] + tc[8];
        float fr = 0.f;
        #pragma unroll
        for (int e = 0; e < 9; ++e) fr += tc[e] * tc[e];
        fr = sqrtf(fr);
        norms[node * D_IN + S_DIM + V_DIM + 2 * c]     = (_Float16)tr;
        norms[node * D_IN + S_DIM + V_DIM + 2 * c + 1] = (_Float16)fr;
    }
    __syncthreads();

    // ---- GEMM1: [16 x 320] @ [320 x 160], 10 N-tiles over 4 waves ----
    for (int nt = wave; nt < NT1; nt += 4) {
        v8f acc = {};
        #pragma unroll
        for (int ks = 0; ks < K1STEPS; ++ks) {
            v8h a0 = *(const v8h*)&norms[col * D_IN + ks * 32 + hi * 8];
            v8h a1 = *(const v8h*)&norms[col * D_IN + ks * 32 + 16 + hi * 8];
            v16h a;
            #pragma unroll
            for (int j = 0; j < 8; ++j) { a[j] = a0[j]; a[j + 8] = a1[j]; }
            v16h b = *(const v16h*)&w1p[(size_t)((nt * K1STEPS + ks) * 32 + lane) * 16];
            acc = __builtin_amdgcn_wmma_f32_16x16x32_f16(false, a, false, b,
                                                         (short)0, acc, false, false);
        }
        float bias = b1[nt * 16 + col];
        #pragma unroll
        for (int r = 0; r < 8; ++r) {
            float x = acc[r] + bias;
            float sx = x / (1.0f + __expf(-x));        // SiLU
            int row = r + hi * 8;                       // node index in tile
            hidden[row * D_MID + nt * 16 + col] = (_Float16)sx;
        }
    }
    __syncthreads();

    // ---- GEMM2: [16 x 160] @ [160 x 64], one N-tile per wave ----
    {
        const int nt = wave;                            // 0..3
        v8f acc = {};
        #pragma unroll
        for (int ks = 0; ks < K2STEPS; ++ks) {
            v8h a0 = *(const v8h*)&hidden[col * D_MID + ks * 32 + hi * 8];
            v8h a1 = *(const v8h*)&hidden[col * D_MID + ks * 32 + 16 + hi * 8];
            v16h a;
            #pragma unroll
            for (int j = 0; j < 8; ++j) { a[j] = a0[j]; a[j + 8] = a1[j]; }
            v16h b = *(const v16h*)&w2p[(size_t)((nt * K2STEPS + ks) * 32 + lane) * 16];
            acc = __builtin_amdgcn_wmma_f32_16x16x32_f16(false, a, false, b,
                                                         (short)0, acc, false, false);
        }
        float bias = b2[nt * 16 + col];
        #pragma unroll
        for (int r = 0; r < 8; ++r) {
            int row = r + hi * 8;
            alphas_s[row * T_DIM + nt * 16 + col] = acc[r] + bias;
        }
    }
    __syncthreads();

    // ---- Phase 3: gated readout from LDS t, symmetrize, rotate, segment-sum ----
    {
        int node = tid >> 3;          // 0..15
        int grp  = tid & 7;           // 8 channel-groups per node
        float accv[9];
        #pragma unroll
        for (int e = 0; e < 9; ++e) accv[e] = 0.f;
        #pragma unroll
        for (int cc = 0; cc < 8; ++cc) {
            int c = grp * 8 + cc;
            float a = alphas_s[node * T_DIM + c];
            const float* tc = &t_lds[node * 576 + c * 9];
            #pragma unroll
            for (int e = 0; e < 9; ++e) accv[e] += a * tc[e];
        }
        #pragma unroll
        for (int e = 0; e < 9; ++e) partial[(node * 8 + grp) * 9 + e] = accv[e];
    }
    __syncthreads();
    if (tid < 16 && nodeBase + tid < Ntot) {
        const int node = tid;
        float tl[9];
        #pragma unroll
        for (int e = 0; e < 9; ++e) tl[e] = 0.f;
        #pragma unroll
        for (int g = 0; g < 8; ++g)
            #pragma unroll
            for (int e = 0; e < 9; ++e) tl[e] += partial[(node * 8 + g) * 9 + e];
        float s9[9];
        #pragma unroll
        for (int a = 0; a < 3; ++a)
            #pragma unroll
            for (int b = 0; b < 3; ++b)
                s9[a * 3 + b] = 0.5f * (tl[a * 3 + b] + tl[b * 3 + a]);
        const float* Rp = frames + (size_t)(nodeBase + node) * 9;
        float R[9];
        #pragma unroll
        for (int e = 0; e < 9; ++e) R[e] = Rp[e];
        float tmp[9];
        #pragma unroll
        for (int a = 0; a < 3; ++a)
            #pragma unroll
            for (int c = 0; c < 3; ++c)
                tmp[a * 3 + c] = R[a * 3 + 0] * s9[0 * 3 + c]
                               + R[a * 3 + 1] * s9[1 * 3 + c]
                               + R[a * 3 + 2] * s9[2 * 3 + c];
        float o9[9];
        #pragma unroll
        for (int a = 0; a < 3; ++a)
            #pragma unroll
            for (int d = 0; d < 3; ++d)
                o9[a * 3 + d] = tmp[a * 3 + 0] * R[d * 3 + 0]
                              + tmp[a * 3 + 1] * R[d * 3 + 1]
                              + tmp[a * 3 + 2] * R[d * 3 + 2];
        long long g = batch[nodeBase + node];
        #pragma unroll
        for (int e = 0; e < 9; ++e)
            atomicAdd(&out[(size_t)g * 9 + e], o9[e]);
    }
}

extern "C" void kernel_launch(void* const* d_in, const int* in_sizes, int n_in,
                              void* d_out, int out_size, void* d_ws, size_t ws_size,
                              hipStream_t stream) {
    const float*     h      = (const float*)d_in[0];
    const float*     v      = (const float*)d_in[1];
    const float*     t      = (const float*)d_in[2];
    const float*     frames = (const float*)d_in[3];
    const long long* batch  = (const long long*)d_in[4];
    const float*     W1     = (const float*)d_in[5];
    const float*     b1     = (const float*)d_in[6];
    const float*     W2     = (const float*)d_in[7];
    const float*     b2     = (const float*)d_in[8];
    float*           out    = (float*)d_out;

    const int Ntot = in_sizes[0] / S_DIM;           // 200000

    _Float16* w1p = (_Float16*)d_ws;                // 320*160 halves = 100 KB
    _Float16* w2p = w1p + D_IN * D_MID;             // 160*64 halves  =  20 KB

    const int w1total = NT1 * K1STEPS * 32 * 16;    // 51200
    const int w2total = NT2 * K2STEPS * 32 * 16;    // 10240
    pack_weights_kernel<<<(w1total + 255) / 256, 256, 0, stream>>>(W1, w1p, D_MID, K1STEPS, w1total);
    pack_weights_kernel<<<(w2total + 255) / 256, 256, 0, stream>>>(W2, w2p, T_DIM, K2STEPS, w2total);
    zero_out_kernel<<<(out_size + 255) / 256, 256, 0, stream>>>(out, out_size);

    const int nTiles = (Ntot + 15) / 16;            // 12500
    tensor_msg_kernel<<<nTiles, 128, 0, stream>>>(h, v, t, frames, batch,
                                                  w1p, b1, w2p, b2, out, Ntot);
}
